// MaskGRU_55628416418418
// MI455X (gfx1250) — compile-verified, matrix operand order checked by reading, not run on hardware
//
#include <hip/hip_runtime.h>
#include <hip/hip_bf16.h>
#include <stdint.h>

// ---------------------------------------------------------------------------
// Masked GRU, T=2048 B=32 D=256 H=256  (gfx1250 / MI455X, wave32, WMMA bf16)
// ---------------------------------------------------------------------------

typedef __bf16 bf16_t;
typedef __attribute__((ext_vector_type(16))) __bf16 v16bf;
typedef __attribute__((ext_vector_type(8)))  float  v8f;
typedef __attribute__((ext_vector_type(4)))  int    v4i;

#define T_STEPS 2048
#define BATCH   32
#define DIM     256
#define HID     256
#define G3      768            // 3*H
#define NT      48             // G3 / 16   (N tiles)
#define KC      8              // 256 / 32  (K chunks per bf16 WMMA)
#define FRAG_ELEMS (NT * KC * 32 * 16)   // packed weight elements (196608)

#if __has_builtin(__builtin_amdgcn_global_load_async_to_lds_b128) && \
    __has_builtin(__builtin_amdgcn_s_wait_asynccnt)
#define USE_ASYNC_GI 1
#else
#define USE_ASYNC_GI 0
#endif

union Frag16 { uint4 u[2]; v16bf v; };

__device__ __forceinline__ v8f wmma_bf16(v16bf a, v16bf b, v8f c) {
  return __builtin_amdgcn_wmma_f32_16x16x32_bf16(
      /*neg_a=*/false, a, /*neg_b=*/false, b,
      /*c_mod=*/(short)0, c, /*reuse_a=*/false, /*reuse_b=*/false);
}

#if USE_ASYNC_GI
// 16-byte async copy global -> LDS (ASYNCcnt-tracked). Builtin takes
// (v4i addrspace(1)*, v4i addrspace(3)*, imm offset, imm cpol). Generic
// pointers are converted via integer casts: global generic == as1 address;
// LDS generic address keeps the LDS offset in its low 32 bits (ISA 10.2).
typedef __attribute__((address_space(1))) v4i g_v4i;
typedef __attribute__((address_space(3))) v4i l_v4i;

__device__ __forceinline__ void async_cp16(const void* g, void* l) {
  g_v4i* gp = (g_v4i*)(unsigned long long)g;
  l_v4i* lp = (l_v4i*)(unsigned int)(unsigned long long)l;
  __builtin_amdgcn_global_load_async_to_lds_b128(gp, lp, 0, 0);
}
#endif

// ---------------------------------------------------------------------------
// Kernel 1: pack w_ih / w_hh (fp32 row-major [g][d]) into bf16 B-matrix
// fragments in per-lane order:  [nTile][kChunk][lane][16 elems]   (ISA 7.12.2)
// ---------------------------------------------------------------------------
__global__ void __launch_bounds__(256)
pack_weights(const float* __restrict__ w_ih, const float* __restrict__ w_hh,
             bf16_t* __restrict__ wih_pack, bf16_t* __restrict__ whh_pack) {
  int idx = blockIdx.x * blockDim.x + threadIdx.x;       // 0 .. 2*FRAG_ELEMS
  int which = (idx >= FRAG_ELEMS) ? 1 : 0;
  int rem   = which ? (idx - FRAG_ELEMS) : idx;
  int i     = rem & 15;
  int lane  = (rem >> 4) & 31;
  int kc    = (rem >> 9) & 7;
  int nTile = rem >> 12;
  int hf    = lane >> 4;
  int g     = nTile * 16 + (lane & 15);
  int o     = (i < 8) ? (8 * hf + i) : (16 + 8 * hf + (i - 8));
  int d     = kc * 32 + o;
  const float* src = which ? w_hh : w_ih;
  bf16_t*      dst = which ? whh_pack : wih_pack;
  dst[rem] = (bf16_t)src[g * DIM + d];
}

// ---------------------------------------------------------------------------
// Kernel 2: gi = seq_embs(65536x256) @ w_ih^T(256x768) + b_ih   (fp32 out)
// One 16x16 tile per wave, 8 waves / block, bf16 WMMA, K chained x8.
// ---------------------------------------------------------------------------
__global__ void __launch_bounds__(256)
gemm_gi(const float* __restrict__ seq, const bf16_t* __restrict__ wih_pack,
        const float* __restrict__ b_ih, float* __restrict__ gi) {
  int lane  = threadIdx.x & 31;
  int wave  = threadIdx.x >> 5;
  int tile  = blockIdx.x * 8 + wave;                     // 0 .. 4096*48-1
  int mTile = tile / NT;
  int nTile = tile - mTile * NT;
  int hf    = lane >> 4;
  int mrow  = lane & 15;

  const float* arow = seq + (size_t)(mTile * 16 + mrow) * DIM;
  v8f acc = {0.f, 0.f, 0.f, 0.f, 0.f, 0.f, 0.f, 0.f};

#pragma unroll
  for (int kc = 0; kc < KC; ++kc) {
    int d0 = kc * 32 + 8 * hf;
    float4 f0 = *(const float4*)(arow + d0);
    float4 f1 = *(const float4*)(arow + d0 + 4);
    float4 f2 = *(const float4*)(arow + d0 + 16);
    float4 f3 = *(const float4*)(arow + d0 + 20);
    v16bf a;
    a[0]  = (bf16_t)f0.x; a[1]  = (bf16_t)f0.y; a[2]  = (bf16_t)f0.z; a[3]  = (bf16_t)f0.w;
    a[4]  = (bf16_t)f1.x; a[5]  = (bf16_t)f1.y; a[6]  = (bf16_t)f1.z; a[7]  = (bf16_t)f1.w;
    a[8]  = (bf16_t)f2.x; a[9]  = (bf16_t)f2.y; a[10] = (bf16_t)f2.z; a[11] = (bf16_t)f2.w;
    a[12] = (bf16_t)f3.x; a[13] = (bf16_t)f3.y; a[14] = (bf16_t)f3.z; a[15] = (bf16_t)f3.w;

    Frag16 b;
    const uint4* bp = (const uint4*)(wih_pack + ((((size_t)nTile * KC + kc) * 32 + lane) << 4));
    b.u[0] = bp[0];
    b.u[1] = bp[1];
    acc = wmma_bf16(a, b.v, acc);
  }

  int   col  = nTile * 16 + mrow;
  float bias = b_ih[col];
  float* crow = gi + (size_t)(mTile * 16 + 8 * hf) * G3 + col;   // C/D: row = r + 8*half
#pragma unroll
  for (int r = 0; r < 8; ++r) crow[(size_t)r * G3] = acc[r] + bias;
}

// ---------------------------------------------------------------------------
// Kernel 3: persistent scan. 2 blocks x 768 threads (24 waves).
// Block owns 16 batch rows. Per step: gh = h @ w_hh^T via bf16 WMMA
// (2 N-tiles per wave, interleaved chains). gi for step t+1 is async-copied
// into an LDS double buffer during step t (ASYNCcnt). Carried fp32 h lives
// in per-thread registers; only the bf16 A-fragment copy lives in LDS.
// ---------------------------------------------------------------------------
#define HB_STRIDE 264      // bf16 h row stride (pad 8 -> conflict-free b128 LDS loads)
#define GH_STRIDE 772      // fp32 gh row stride (rows r / r+8 land 32 banks apart)
#define EPT 6              // max gate elements per thread (ceil(4096/768))

__global__ void __launch_bounds__(768)
gru_scan(const float* __restrict__ gi, const int* __restrict__ mask,
         const float* __restrict__ hidden, const bf16_t* __restrict__ whh_pack,
         const float* __restrict__ b_hh, float* __restrict__ out) {
  __shared__ float  ghL[16 * GH_STRIDE];
  __shared__ bf16_t hB [16 * HB_STRIDE];
  __shared__ float  bhh[G3];
#if USE_ASYNC_GI
  __shared__ float  giL[2][16 * G3];     // 2 x 48KB double buffer
#endif

  const int tid  = threadIdx.x;
  const int lane = tid & 31;
  const int wave = tid >> 5;                 // 0..23
  const int bOff = blockIdx.x * 16;          // batch rows [bOff, bOff+16)
  const int hf   = lane >> 4;
  const int mrow = lane & 15;
  const int n0   = wave * 2;
  const int n1   = wave * 2 + 1;

  // carried fp32 hidden state: fixed (b,j) ownership per thread, in registers
  float hreg[EPT];
#pragma unroll
  for (int i = 0; i < EPT; ++i) {
    int e = tid + i * 768;
    if (e < 16 * HID) {
      int b = e >> 8, j = e & 255;
      float h0 = hidden[(size_t)(bOff + b) * HID + j];
      hreg[i] = h0;
      hB[b * HB_STRIDE + j] = (bf16_t)h0;
    }
  }
  if (tid < G3) bhh[tid] = b_hh[tid];

  const float* giBlk = gi + (size_t)bOff * G3;     // rows for this block, step 0
#if USE_ASYNC_GI
  // stage gi(t=0) into buffer 0: 16x768 fp32 rows are contiguous -> flat copy
#pragma unroll
  for (int u = 0; u < 4; ++u) {
    int idx = u * 768 + tid;                       // 3072 x 16B transfers
    async_cp16((const char*)giBlk + (size_t)idx * 16, (char*)&giL[0][0] + idx * 16);
  }
#endif
  __syncthreads();

  for (int t = 0; t < T_STEPS; ++t) {
#if USE_ASYNC_GI
    // kick off async staging of gi(t+1) into the other buffer
    if (t + 1 < T_STEPS) {
      const char* src = (const char*)(giBlk + (size_t)(t + 1) * BATCH * G3);
      char* dst = (char*)&giL[(t + 1) & 1][0];
#pragma unroll
      for (int u = 0; u < 4; ++u) {
        int idx = u * 768 + tid;
        async_cp16(src + (size_t)idx * 16, dst + idx * 16);
      }
    }
#endif

    // ---- gh = h @ w_hh^T : two independent WMMA chains per wave ----
    v8f acc0 = {0.f, 0.f, 0.f, 0.f, 0.f, 0.f, 0.f, 0.f};
    v8f acc1 = {0.f, 0.f, 0.f, 0.f, 0.f, 0.f, 0.f, 0.f};
#pragma unroll
    for (int kc = 0; kc < KC; ++kc) {
      int d0 = kc * 32 + 8 * hf;
      Frag16 a;
      a.u[0] = *(const uint4*)&hB[mrow * HB_STRIDE + d0];
      a.u[1] = *(const uint4*)&hB[mrow * HB_STRIDE + d0 + 16];
      Frag16 b0, b1;
      const uint4* p0 = (const uint4*)(whh_pack + ((((size_t)n0 * KC + kc) * 32 + lane) << 4));
      const uint4* p1 = (const uint4*)(whh_pack + ((((size_t)n1 * KC + kc) * 32 + lane) << 4));
      b0.u[0] = p0[0]; b0.u[1] = p0[1];
      b1.u[0] = p1[0]; b1.u[1] = p1[1];
      acc0 = wmma_bf16(a.v, b0.v, acc0);
      acc1 = wmma_bf16(a.v, b1.v, acc1);
    }
    {
      float* g0 = &ghL[(8 * hf) * GH_STRIDE + n0 * 16 + mrow];
      float* g1 = &ghL[(8 * hf) * GH_STRIDE + n1 * 16 + mrow];
#pragma unroll
      for (int r = 0; r < 8; ++r) {
        g0[r * GH_STRIDE] = acc0[r];
        g1[r * GH_STRIDE] = acc1[r];
      }
    }
#if USE_ASYNC_GI
    // retire gi(t): <=4 outstanding leaves only the just-issued t+1 transfers
    __builtin_amdgcn_s_wait_asynccnt(4);
#endif
    __syncthreads();

    // ---- fused gates ----
#if USE_ASYNC_GI
    const float* giRow = &giL[t & 1][0];
#else
    const float* giRow = giBlk + (size_t)t * BATCH * G3;
#endif
    const int* mrw = mask + t * BATCH + bOff;
#pragma unroll
    for (int i = 0; i < EPT; ++i) {
      int e = tid + i * 768;
      if (e < 16 * HID) {
        int b = e >> 8, j = e & 255;
        const float* gib = giRow + (size_t)b * G3;
        float ir = gib[j], iz = gib[j + 256], inn = gib[j + 512];
        float hr = ghL[b * GH_STRIDE + j]       + bhh[j];
        float hz = ghL[b * GH_STRIDE + j + 256] + bhh[j + 256];
        float hn = ghL[b * GH_STRIDE + j + 512] + bhh[j + 512];
        float r  = 1.f / (1.f + __expf(-(ir + hr)));
        float z  = 1.f / (1.f + __expf(-(iz + hz)));
        float pe = inn + r * hn;
        float e2 = __expf(-2.f * pe);
        float n  = (1.f - e2) / (1.f + e2);
        float hp = hreg[i];
        float ht = (1.f - z) * n + z * hp;
        int   mb = mrw[b];
        float hnew = mb ? ht : hp;
        out[(size_t)t * BATCH * HID + (size_t)(bOff + b) * HID + j] = mb ? ht : 0.f;
        hreg[i] = hnew;
        hB[b * HB_STRIDE + j] = (bf16_t)hnew;
#if !USE_ASYNC_GI
        if (t + 1 < T_STEPS)                     // warm L2 for next step's gi
          __builtin_prefetch(gib + (size_t)BATCH * G3 + j, 0, 1);
#endif
      }
    }
    __syncthreads();
  }

  // h_final appended after outputs
#pragma unroll
  for (int i = 0; i < EPT; ++i) {
    int e = tid + i * 768;
    if (e < 16 * HID) {
      int b = e >> 8, j = e & 255;
      out[(size_t)T_STEPS * BATCH * HID + (size_t)(bOff + b) * HID + j] = hreg[i];
    }
  }
}

// ---------------------------------------------------------------------------
extern "C" void kernel_launch(void* const* d_in, const int* in_sizes, int n_in,
                              void* d_out, int out_size, void* d_ws, size_t ws_size,
                              hipStream_t stream) {
  const float* seq    = (const float*)d_in[0];   // (T,B,D)
  const int*   mask   = (const int*)  d_in[1];   // (T,B)
  const float* hidden = (const float*)d_in[2];   // (B,H)
  const float* w_ih   = (const float*)d_in[3];   // (3H,D)
  const float* w_hh   = (const float*)d_in[4];   // (3H,H)
  const float* b_ih   = (const float*)d_in[5];   // (3H)
  const float* b_hh   = (const float*)d_in[6];   // (3H)
  float* out = (float*)d_out;                    // (T,B,H) ++ (B,H)

  const size_t giElems = (size_t)T_STEPS * BATCH * G3;     // 50.3M fp32
  float*  gi       = (float*)d_ws;
  bf16_t* wih_pack = (bf16_t*)((char*)d_ws + giElems * sizeof(float));
  bf16_t* whh_pack = wih_pack + FRAG_ELEMS;

  pack_weights<<<(2 * FRAG_ELEMS) / 256, 256, 0, stream>>>(w_ih, w_hh, wih_pack, whh_pack);
  gemm_gi<<<(4096 * NT) / 8, 256, 0, stream>>>(seq, wih_pack, b_ih, gi);
  gru_scan<<<2, 768, 0, stream>>>(gi, mask, hidden, whh_pack, b_hh, out);
}
